// LSTM_72679436583235
// MI455X (gfx1250) — matrix operand order repaired and checked by reading, not compiled, not run on hardware
//
#include <hip/hip_runtime.h>
#include <math.h>

// ---------------------------------------------------------------------------
// LSTM forward for MI455X (gfx1250, wave32, WMMA 16x16x32 bf16 -> f32 acc)
//
//  B=64 batches, S=2048 steps, I=H=256, gates=4H=1024.
//  Phase 0: pack W_ih/W_hh -> bf16 in WMMA B-fragment order; biasSum=b_ih+b_hh
//  Phase 1: x_proj = x @ W_ih^T + bias, stored bf16 in C-fragment order
//  Phase 2: 4 persistent WGs (16 batches each) run the sequential scan with
//           gate-aligned wave tiling so i/f/g/o of a cell live in one lane.
//           Recurrence is software-pipelined (B frags prefetched one k-group
//           ahead), xp loads hoisted above the GEMM, h double-buffered in LDS
//           so only one barrier per timestep sits on the serial chain.
// ---------------------------------------------------------------------------

typedef __attribute__((ext_vector_type(16))) __bf16         v16bf;
typedef __attribute__((ext_vector_type(16))) unsigned short v16u;
typedef __attribute__((ext_vector_type(8)))  float          v8f;

#define LB   64          // batch
#define LS   2048        // seq len
#define LH   256         // hidden = input
#define LG   1024        // 4*H
#define OUTS (LB * LS * LH)   // 33,554,432 floats of `outputs`

__device__ __forceinline__ unsigned short f2bf(float f) {
  unsigned u = __builtin_bit_cast(unsigned, f);
  u += 0x7FFFu + ((u >> 16) & 1u);          // round-to-nearest-even
  return (unsigned short)(u >> 16);
}
__device__ __forceinline__ float bf2f(unsigned short h) {
  return __builtin_bit_cast(float, ((unsigned)h) << 16);
}
__device__ __forceinline__ float sigmoidf(float x) {
  return 1.0f / (1.0f + expf(-x));
}

typedef union { uint4 q[2]; v16u v; } frag_u;

// ---------------------------------------------------------------------------
// Phase 0: weight pack.  B-fragment layout (bf16, 32x16 K x N tile, wave32):
//   lane L: n = ntile*16 + (L&15);  k = ktile*32 + ((L&16)?8:0) + (t&7) + ((t&8)?16:0), t=0..15
//   packed[((ntile*8+ktile)*32 + L)*16 + t]  -> per-lane 32B contiguous.
// B[k][n] = W[n][k]  (gates = act @ W^T).
// ---------------------------------------------------------------------------
__global__ __launch_bounds__(256) void lstm_pack(
    const float* __restrict__ W_ih, const float* __restrict__ W_hh,
    const float* __restrict__ b_ih, const float* __restrict__ b_hh,
    unsigned short* __restrict__ wih_p, unsigned short* __restrict__ whh_p,
    float* __restrict__ biasSum) {
  int e = blockIdx.x * 256 + threadIdx.x;
  if (e < LG) biasSum[e] = b_ih[e] + b_hh[e];
  if (e < LG * LH) {
    int t     = e & 15;
    int lane  = (e >> 4) & 31;
    int tile  = e >> 9;
    int kt    = tile & 7;
    int ntile = tile >> 3;
    int n = ntile * 16 + (lane & 15);
    int k = kt * 32 + ((lane & 16) ? 8 : 0) + (t & 7) + ((t & 8) ? 16 : 0);
    wih_p[e] = f2bf(W_ih[n * LH + k]);
    whh_p[e] = f2bf(W_hh[n * LH + k]);
  }
}

// A-fragment (16x32, bf16) load from an LDS tile laid out [m][k] bf16,
// m=0..15, k contiguous (256 per row).  lane L: m=L&15, k-offset lo8=(L&16?8:0)
// halves 0..7 -> k0..k0+7 ; halves 8..15 -> k0+16..k0+23 -> two b128 loads.
__device__ __forceinline__ v16bf load_afrag(const unsigned short* lds,
                                            int mrow, int kt, int lo8) {
  frag_u fa;
  const unsigned short* ap = lds + (mrow * 256 + kt * 32 + lo8);
  fa.q[0] = *(const uint4*)(ap);
  fa.q[1] = *(const uint4*)(ap + 16);
  return __builtin_bit_cast(v16bf, fa.v);
}

// B-fragment flat offset in a packed weight buffer (in ushorts)
__device__ __forceinline__ size_t bfrag_off(int ntile, int kt, int lane) {
  return ((size_t)(ntile * 8 + kt) * 32 + lane) * 16;
}

// ---------------------------------------------------------------------------
// Phase 1: x_proj.  Block = (t, wg, qhalf); A tile = 16 batches at step t
// (matches recurrence row tiling).  Wave q computes gate-aligned ntiles
// {q, q+16, q+32, q+48}; result+bias stored bf16 in C-fragment order:
//   xp[((((wg*S+t)*16 + q)*32 + lane)*4 + gate)*8 + r]
// ---------------------------------------------------------------------------
__global__ __launch_bounds__(256) void lstm_xproj(
    const float* __restrict__ x, const unsigned short* __restrict__ wih_p,
    const float* __restrict__ biasSum, unsigned short* __restrict__ xp) {
  __shared__ unsigned short xlds[16 * 256];
  const int t   = blockIdx.x;
  const int wg  = blockIdx.y;
  const int qh  = blockIdx.z;
  const int tid = threadIdx.x;

  // stage 16 batch rows of x (timestep t) into LDS as bf16
  {
    int rr = tid >> 4;
    int c0 = (tid & 15) * 16;
    const float* xr = x + (((size_t)(wg * 16 + rr) * LS + t) * LH + c0);
#pragma unroll
    for (int ii = 0; ii < 16; ii += 4) {
      float4 v = *(const float4*)(xr + ii);
      xlds[rr * 256 + c0 + ii + 0] = f2bf(v.x);
      xlds[rr * 256 + c0 + ii + 1] = f2bf(v.y);
      xlds[rr * 256 + c0 + ii + 2] = f2bf(v.z);
      xlds[rr * 256 + c0 + ii + 3] = f2bf(v.w);
    }
  }
  __syncthreads();

  const int lane = tid & 31;
  const int q    = (tid >> 5) + qh * 8;
  const int mrow = lane & 15;
  const int lo8  = (lane & 16) ? 8 : 0;

  v8f acc[4] = {};
#pragma unroll
  for (int kt = 0; kt < 8; ++kt) {
    v16bf a = load_afrag(xlds, mrow, kt, lo8);
#pragma unroll
    for (int g = 0; g < 4; ++g) {
      v16u bv = *(const v16u*)(wih_p + bfrag_off(g * 16 + q, kt, lane));
      v16bf b = __builtin_bit_cast(v16bf, bv);
      acc[g] = __builtin_amdgcn_wmma_f32_16x16x32_bf16(
          false, a, false, b, (short)0, acc[g], false, false);
    }
  }

  size_t base = (((size_t)(wg * LS + t) * 16 + q) * 32 + lane) * 4 * 8;
#pragma unroll
  for (int g = 0; g < 4; ++g) {
    float bias = biasSum[g * LH + q * 16 + mrow];
    union { unsigned short s[8]; uint4 u; } pk;
#pragma unroll
    for (int r = 0; r < 8; ++r) pk.s[r] = f2bf(acc[g][r] + bias);
    *(uint4*)(xp + base + (size_t)g * 8) = pk.u;
  }
}

// ---------------------------------------------------------------------------
// Phase 2: recurrence.  4 WGs x 512 threads (16 waves).  WG owns batches
// [wg*16, wg*16+16).  Wave q owns hidden units [16q,16q+16) and all 4 gates
// for them, so the LSTM cell update is lane-local.  c in VGPRs; h ping-pongs
// between two 8KB LDS buffers (one barrier per step); W_hh streamed from L2
// each step (512KB, always L2-resident), prefetched one k-group ahead.
// ---------------------------------------------------------------------------
__global__ __launch_bounds__(512) void lstm_rec(
    const unsigned short* __restrict__ whh_p,
    const unsigned short* __restrict__ xp,
    float* __restrict__ out) {
  __shared__ unsigned short hlds[2][16 * 256];  // h as bf16, [m][j], ping-pong
  const int tid  = threadIdx.x;
  const int wg   = blockIdx.x;
  const int lane = tid & 31;
  const int q    = tid >> 5;                  // 0..15 hidden block
  const int mrow = lane & 15;
  const int lo8  = (lane & 16) ? 8 : 0;       // A k-offset AND C row offset
  const int j    = q * 16 + mrow;             // hidden unit this lane owns

  for (int i = tid; i < 16 * 256; i += 512) hlds[0][i] = 0;   // h0 = 0

  float cst[8];                               // c state, rows r / r+8
#pragma unroll
  for (int r = 0; r < 8; ++r) cst[r] = 0.0f;

  __syncthreads();

  for (int t = 0; t < LS; ++t) {
    const unsigned short* hbuf = hlds[t & 1];
    unsigned short*       hnxt = hlds[(t + 1) & 1];

    // xp (+bias, C-fragment order) is independent of h: issue these global
    // loads first so their latency overlaps the entire GEMM below.
    size_t base = (((size_t)(wg * LS + t) * 16 + q) * 32 + lane) * 4 * 8;
    union { uint4 u[4]; unsigned short s[32]; } xf;
#pragma unroll
    for (int g = 0; g < 4; ++g) xf.u[g] = *(const uint4*)(xp + base + (size_t)g * 8);

    // gates_partial = h(t) @ W_hh^T, software-pipelined one k-group ahead
    v8f acc[4] = {};
    v16bf a_cur = load_afrag(hbuf, mrow, 0, lo8);
    v16u  b_cur[4];
#pragma unroll
    for (int g = 0; g < 4; ++g)
      b_cur[g] = *(const v16u*)(whh_p + bfrag_off(g * 16 + q, 0, lane));

#pragma unroll
    for (int kt = 0; kt < 8; ++kt) {
      v16bf a_nxt;
      v16u  b_nxt[4];
      if (kt < 7) {                       // prefetch next k-group
        a_nxt = load_afrag(hbuf, mrow, kt + 1, lo8);
#pragma unroll
        for (int g = 0; g < 4; ++g)
          b_nxt[g] = *(const v16u*)(whh_p + bfrag_off(g * 16 + q, kt + 1, lane));
      }
#pragma unroll
      for (int g = 0; g < 4; ++g) {
        v16bf b = __builtin_bit_cast(v16bf, b_cur[g]);
        acc[g] = __builtin_amdgcn_wmma_f32_16x16x32_bf16(
            false, a_cur, false, b, (short)0, acc[g], false, false);
      }
      if (kt < 7) {
        a_cur = a_nxt;
#pragma unroll
        for (int g = 0; g < 4; ++g) b_cur[g] = b_nxt[g];
      }
    }

    // lane-local LSTM cell update for 8 (batch-row, hidden-j) cells
#pragma unroll
    for (int r = 0; r < 8; ++r) {
      float iv = sigmoidf(acc[0][r] + bf2f(xf.s[0 * 8 + r]));
      float fv = sigmoidf(acc[1][r] + bf2f(xf.s[1 * 8 + r]));
      float gv = tanhf  (acc[2][r] + bf2f(xf.s[2 * 8 + r]));
      float ov = sigmoidf(acc[3][r] + bf2f(xf.s[3 * 8 + r]));
      float cn = fv * cst[r] + iv * gv;
      cst[r]   = cn;
      float hn = ov * tanhf(cn);

      int m = r + lo8;                 // C-layout row within the 16-row tile
      int b = wg * 16 + m;
      out[((size_t)b * LS + t) * LH + j] = hn;          // outputs[b,t,j]
      hnxt[m * 256 + j] = f2bf(hn);                     // h for step t+1
      if (t == LS - 1) {
        out[OUTS + (size_t)b * LH + j]            = hn; // final h
        out[OUTS + LB * LH + (size_t)b * LH + j]  = cn; // final c
      }
    }
    // single barrier per step: h(t+1) fully written before anyone reads it;
    // writes to hnxt cannot race reads of hbuf (different buffers).
    __syncthreads();
  }
}

// ---------------------------------------------------------------------------
extern "C" void kernel_launch(void* const* d_in, const int* in_sizes, int n_in,
                              void* d_out, int out_size, void* d_ws, size_t ws_size,
                              hipStream_t stream) {
  (void)in_sizes; (void)n_in; (void)out_size; (void)ws_size;
  const float* x    = (const float*)d_in[0];
  const float* W_ih = (const float*)d_in[1];
  const float* W_hh = (const float*)d_in[2];
  const float* b_ih = (const float*)d_in[3];
  const float* b_hh = (const float*)d_in[4];

  // workspace layout:
  //   [0, 512K)        W_ih packed bf16
  //   [512K, 1M)       W_hh packed bf16
  //   [1M, 1M+4K)      biasSum f32
  //   [1M+64K, +256M)  x_proj bf16, C-fragment order
  char* ws = (char*)d_ws;
  unsigned short* wih_p  = (unsigned short*)ws;
  unsigned short* whh_p  = (unsigned short*)(ws + (512u << 10));
  float*          biasSm = (float*)(ws + (1u << 20));
  unsigned short* xp     = (unsigned short*)(ws + (1u << 20) + (64u << 10));

  lstm_pack <<<dim3(1024),        dim3(256), 0, stream>>>(W_ih, W_hh, b_ih, b_hh,
                                                          wih_p, whh_p, biasSm);
  lstm_xproj<<<dim3(2048, 4, 2),  dim3(256), 0, stream>>>(x, wih_p, biasSm, xp);
  lstm_rec  <<<dim3(4),           dim3(512), 0, stream>>>(whh_p, xp, (float*)d_out);
}
